// PCNN_ONE_10282151707669
// MI455X (gfx1250) — matrix operand order repaired and checked by reading, not compiled, not run on hardware
//
#include <hip/hip_runtime.h>
#include <hip/hip_bf16.h>

typedef __attribute__((ext_vector_type(16))) _Float16 v16h;
typedef __attribute__((ext_vector_type(8)))  float    v8f;
typedef __attribute__((ext_vector_type(4)))  int      v4i_;

#define DW_    50
#define DP_    5
#define D_     60
#define OC_    230
#define NREL_  53
#define LIMIT_ 30
#define G_     1000
#define B_     8
#define L_     120
#define N_     (G_*B_)

// Padded GEMM dims: K laid out as kh*64 + d (d in [0,64), cols 60..63 zero)
#define KP   192   // 3 * 64
#define OCP  240   // 15 * 16
#define RX   132   // x_pad rows: row r = token r-1; rows 0 and 121..131 are zero

#define NEGF (-3.0e38f)

#if __has_builtin(__builtin_amdgcn_global_load_async_to_lds_b128)
#define HAVE_ASYNC_LDS 1
#else
#define HAVE_ASYNC_LDS 0
#endif

// ---------------------------------------------------------------- init output
__global__ void init_out_kernel(unsigned* __restrict__ out) {
    int i = blockIdx.x * blockDim.x + threadIdx.x;
    if (i < G_ * NREL_) out[i] = 0xFF800000u;   // -inf bits; reduced via unsigned atomicMin
}

// ------------------------------------------------- conv weights -> fp16 [oc][k]
__global__ void prep_w_kernel(const float* __restrict__ convw, _Float16* __restrict__ Wh) {
    int i = blockIdx.x * blockDim.x + threadIdx.x;
    if (i >= OCP * KP) return;
    int oc = i / KP;
    int kk = i % KP;
    int kh = kk >> 6;
    int d  = kk & 63;
    float v = 0.f;
    if (oc < OC_ && d < D_) v = convw[(oc * 3 + kh) * D_ + d];  // OIHW flat: oc*180 + kh*60 + d
    Wh[i] = (_Float16)v;
}

// ------------------------------------------------------------------ main kernel
__launch_bounds__(256)
__global__ void pcnn_main_kernel(const int*   __restrict__ words,
                                 const int*   __restrict__ pos,
                                 const float* __restrict__ w2v,
                                 const float* __restrict__ p1e,
                                 const float* __restrict__ p2e,
                                 const float* __restrict__ cb0,
                                 const float* __restrict__ cb1,
                                 const float* __restrict__ cb2,
                                 const float* __restrict__ remb,
                                 const float* __restrict__ rbias,
                                 const _Float16* __restrict__ Wh,
                                 unsigned*    __restrict__ out) {
    __shared__ _Float16 wt[OCP * KP];       // 92160 B, oc-major weights
    __shared__ _Float16 xp[RX * 64];        // 16896 B, padded sentence
    __shared__ float    pool[8 * 3 * OCP];  // per-wave piecewise maxima
    __shared__ int      widx[L_], pidx0[L_], pidx1[L_], seg[L_];
    __shared__ int      cnt[3];
    __shared__ float    flds[3 * OC_];
    __shared__ float    lg[NREL_];
    __shared__ float    lse_s;

    const int t    = threadIdx.x;
    const int n    = blockIdx.x;          // sentence index
    const int lane = t & 31;
    const int w    = t >> 5;              // wave id = M-tile id

    // Phase 0: zero piece counters; stage weights into LDS.
    // CDNA5 async global->LDS DMA (ASYNCcnt path, no VGPR round-trip);
    // falls back to vector load/store copies if the builtin is unavailable.
    if (t < 3) cnt[t] = 0;
    {
        const uint4* src = (const uint4*)Wh;
        uint4*       dst = (uint4*)wt;
#if HAVE_ASYNC_LDS
        for (int i = t; i < (OCP * KP) / 8; i += 256) {
            __builtin_amdgcn_global_load_async_to_lds_b128(
                (__attribute__((address_space(1))) v4i_*)(src + i),
                (__attribute__((address_space(3))) v4i_*)(dst + i),
                0, 0);
        }
#else
        for (int i = t; i < (OCP * KP) / 8; i += 256) dst[i] = src[i];
#endif
    }

    // Phase 1: token indices, piece segmentation, piece counts
    if (t < L_) {
        int wl = words[n * L_ + t];
        int p0 = pos[(n * L_ + t) * 2 + 0];
        int p1 = pos[(n * L_ + t) * 2 + 1];
        widx[t] = wl; pidx0[t] = p0; pidx1[t] = p1;
        int b0 = (p0 >= LIMIT_) ? 1 : 0;
        int b1 = (p1 >= LIMIT_) ? 1 : 0;
        int s  = (b0 & b1) ? 0 : (((b0 | b1) == 0) ? 2 : 1);   // 0=left 1=mid 2=right
        seg[t] = s;
        atomicAdd(&cnt[s], 1);
    }

#if HAVE_ASYNC_LDS
  #if __has_builtin(__builtin_amdgcn_s_wait_asynccnt)
    __builtin_amdgcn_s_wait_asynccnt(0);
  #else
    asm volatile("s_wait_asynccnt 0x0" ::: "memory");
  #endif
#endif
    __syncthreads();

    // Phase 2: gather embeddings -> fp16 x_pad[r][d] (r=0 & r>120 are zero halo)
    for (int i = t; i < RX * 64; i += 256) {
        int r = i >> 6, d = i & 63;
        float v = 0.f;
        if (r >= 1 && r <= L_) {
            int l = r - 1;
            if (d < DW_)            v = w2v[widx[l] * DW_ + d];
            else if (d < DW_ + DP_) v = p1e[pidx0[l] * DP_ + (d - DW_)];
            else if (d < D_)        v = p2e[pidx1[l] * DP_ + (d - DW_ - DP_)];
        }
        xp[i] = (_Float16)v;
    }
    __syncthreads();

    // Phase 3: GEMM via WMMA. Wave w owns output rows l = 16w..16w+15.
    union U { uint4 u[2]; v16h h; };
    const int l0   = w * 16;
    const int mrow = lane & 15;
    const int hi   = lane >> 4;

    // Preload the 6 A fragments (16x32 f16 per ISA layout):
    // lane: halves 0..7 hold k = kt*32 + hi*8 + (0..7); halves 8..15 hold +16.
    v16h a[6];
#pragma unroll
    for (int kt = 0; kt < 6; ++kt) {
        int ks1 = kt * 32 + hi * 8;
        int ks2 = ks1 + 16;
        int r1  = l0 + mrow + (ks1 >> 6);   // row = l + kh   (im2col fold)
        int r2  = l0 + mrow + (ks2 >> 6);
        U tmp;
        tmp.u[0] = *(const uint4*)(xp + r1 * 64 + (ks1 & 63));
        tmp.u[1] = *(const uint4*)(xp + r2 * 64 + (ks2 & 63));
        a[kt] = tmp.h;
    }

    // Hoisted piecewise masks: bit v of bm{p} set iff row m = l0+8*hi+v is a
    // valid sentence position belonging to piece p. (seg is nt-invariant.)
    int bm0 = 0, bm1 = 0, bm2 = 0;
#pragma unroll
    for (int v = 0; v < 8; ++v) {
        int m  = l0 + hi * 8 + v;
        int mc = (m < L_) ? m : (L_ - 1);
        int s  = seg[mc];
        int ok = (m < L_) ? 1 : 0;
        bm0 |= (ok & (s == 0 ? 1 : 0)) << v;
        bm1 |= (ok & (s == 1 ? 1 : 0)) << v;
        bm2 |= (ok & (s == 2 ? 1 : 0)) << v;
    }

    const _Float16* bbase = wt + (lane & 15) * KP + hi * 8;
    for (int nt = 0; nt < 15; ++nt) {
        // B fragments: 2x ds_load_b128 per kt; scheduler overlaps each pair
        // with the preceding WMMA in the chain.
        U tb[6];
#pragma unroll
        for (int kt = 0; kt < 6; ++kt) {
            const uint4* bp = (const uint4*)(bbase + nt * 16 * KP + kt * 32);
            tb[kt].u[0] = bp[0];
            tb[kt].u[1] = bp[2];
        }
        v8f c = {};
#pragma unroll
        for (int kt = 0; kt < 6; ++kt) {
            c = __builtin_amdgcn_wmma_f32_16x16x32_f16(false, a[kt], false, tb[kt].h,
                                                       (short)0, c, false, false);
        }
        // Branchless fused piecewise max-pool epilogue (pure VALU, co-executes
        // with the matrix pipe; no exec-mask branches).
        float pm0 = NEGF, pm1 = NEGF, pm2 = NEGF;
#pragma unroll
        for (int v = 0; v < 8; ++v) {
            float val = c[v];
            pm0 = fmaxf(pm0, ((bm0 >> v) & 1) ? val : NEGF);
            pm1 = fmaxf(pm1, ((bm1 >> v) & 1) ? val : NEGF);
            pm2 = fmaxf(pm2, ((bm2 >> v) & 1) ? val : NEGF);
        }
        pm0 = fmaxf(pm0, __shfl_xor(pm0, 16, 32));
        pm1 = fmaxf(pm1, __shfl_xor(pm1, 16, 32));
        pm2 = fmaxf(pm2, __shfl_xor(pm2, 16, 32));
        if (hi == 0) {
            int oc = nt * 16 + (lane & 15);
            pool[(w * 3 + 0) * OCP + oc] = pm0;
            pool[(w * 3 + 1) * OCP + oc] = pm1;
            pool[(w * 3 + 2) * OCP + oc] = pm2;
        }
    }
    __syncthreads();

    // Phase 4: reduce over waves; replicate jnp.max(feat*mask) semantics
    // (zeros participate whenever a piece doesn't cover all 120 positions),
    // add per-piece bias, tanh.
    for (int j = t; j < 3 * OC_; j += 256) {
        int p = j / OC_, oc = j % OC_;
        float mx = NEGF;
#pragma unroll
        for (int ww = 0; ww < 8; ++ww) mx = fmaxf(mx, pool[(ww * 3 + p) * OCP + oc]);
        if (cnt[p] < L_) mx = fmaxf(mx, 0.f);
        const float* cb = (p == 0) ? cb0 : ((p == 1) ? cb1 : cb2);
        flds[j] = tanhf(mx + cb[oc]);
    }
    __syncthreads();

    // Phase 5: classifier (690 -> 53), 4 threads per relation + shuffle reduce;
    // eval-mode 0.5 scale + bias.
    if (t < 4 * NREL_) {
        int rel = t >> 2;
        int q   = t & 3;
        const float* re = remb + rel * (3 * OC_);
        float acc = 0.f;
        for (int j = q; j < 3 * OC_; j += 4) acc += flds[j] * re[j];
        acc += __shfl_xor(acc, 1, 32);
        acc += __shfl_xor(acc, 2, 32);
        if (q == 0) lg[rel] = acc * 0.5f + rbias[rel];
    }
    __syncthreads();
    if (t == 0) {
        float m = NEGF;
        for (int i = 0; i < NREL_; ++i) m = fmaxf(m, lg[i]);
        float s = 0.f;
        for (int i = 0; i < NREL_; ++i) s += expf(lg[i] - m);
        lse_s = m + logf(s);
    }
    __syncthreads();

    // Phase 6: log-softmax + bag-max. logp <= 0, so for non-positive floats
    // unsigned-min of the bit pattern == float max. d_out pre-init to -inf bits.
    if (t < NREL_) {
        float lp = lg[t] - lse_s;
        int   g  = n / B_;
        atomicMin(&out[g * NREL_ + t], __float_as_uint(lp));
    }
}

extern "C" void kernel_launch(void* const* d_in, const int* in_sizes, int n_in,
                              void* d_out, int out_size, void* d_ws, size_t ws_size,
                              hipStream_t stream) {
    const int*   words = (const int*)d_in[0];
    const int*   pos   = (const int*)d_in[1];
    const float* w2v   = (const float*)d_in[2];
    const float* p1e   = (const float*)d_in[3];
    const float* p2e   = (const float*)d_in[4];
    const float* convw = (const float*)d_in[5];
    const float* cb0   = (const float*)d_in[6];
    const float* cb1   = (const float*)d_in[7];
    const float* cb2   = (const float*)d_in[8];
    const float* remb  = (const float*)d_in[9];
    const float* rbias = (const float*)d_in[10];

    _Float16* Wh  = (_Float16*)d_ws;        // 92160 B scratch for fp16 weights
    unsigned* out = (unsigned*)d_out;

    init_out_kernel<<<(G_ * NREL_ + 255) / 256, 256, 0, stream>>>(out);
    prep_w_kernel<<<(OCP * KP + 255) / 256, 256, 0, stream>>>(convw, Wh);
    pcnn_main_kernel<<<N_, 256, 0, stream>>>(words, pos, w2v, p1e, p2e,
                                             cb0, cb1, cb2, remb, rbias,
                                             (const _Float16*)Wh, out);
}
